// AttentiveConv3d_77163382440397
// MI455X (gfx1250) — matrix operand order repaired and checked by reading, not compiled, not run on hardware
//
#include <hip/hip_runtime.h>

typedef _Float16 v4h  __attribute__((ext_vector_type(4)));
typedef _Float16 v8h  __attribute__((ext_vector_type(8)));
typedef _Float16 v16h __attribute__((ext_vector_type(16)));
typedef float    v8f  __attribute__((ext_vector_type(8)));

#define B_   2
#define C_   128
#define T_   16
#define H_   28
#define W_   28
#define HW_  (H_*W_)      // 784
#define THW_ (T_*H_*W_)   // 12544  (divisible by 16 -> M-tiles never straddle batch)
#define P_   (B_*THW_)    // 25088 = 1568 * 16
#define K27  27

// ---------------- Kernel 1: attention -> merged f16 [P][C] ----------------
// block = (32 w-lanes, 8 channel-threads); grid = B*T*H rows
__global__ __launch_bounds__(256) void attn_kernel(const float* __restrict__ x,
                                                   const float* __restrict__ q,
                                                   _Float16* __restrict__ Mrg) {
    __shared__ float part[8][K27][32];
    __shared__ float attn[2][K27][32];

    const int bid = blockIdx.x;
    const int hr  = bid % H_;
    const int t   = (bid / H_) % T_;
    const int b   = bid / (H_ * T_);
    const int w   = threadIdx.x;             // 0..31 (lanes 28..31 compute discarded work)
    const int cy  = threadIdx.y;             // 0..7
    const int wl  = (w < W_) ? w : (W_ - 1); // clamp for safe addressing

    // ---- stage 1: partial logits over this thread's 16 channels ----
    float acc[K27];
    #pragma unroll
    for (int k = 0; k < K27; ++k) acc[k] = 0.f;

    for (int j = 0; j < 16; ++j) {
        const int c = cy + 8 * j;                       // parity(c) == parity(cy) == head
        const float qc = q[(c & 1) * 64 + (c >> 1)] * (1.0f / 64.0f); // scale on q AND k
        const float* xc = x + (size_t)(b * C_ + c) * THW_;
        int k = 0;
        #pragma unroll
        for (int dt = -1; dt <= 1; ++dt) {
            const int tt = t + dt; const bool okt = (unsigned)tt < (unsigned)T_;
            #pragma unroll
            for (int dh = -1; dh <= 1; ++dh) {
                const int hh = hr + dh; const bool okh = (unsigned)hh < (unsigned)H_;
                #pragma unroll
                for (int dw = -1; dw <= 1; ++dw) {
                    const int ww = wl + dw; const bool okw = (unsigned)ww < (unsigned)W_;
                    const float v = (okt & okh & okw) ? xc[(tt * H_ + hh) * W_ + ww] : 0.f;
                    acc[k] += qc * v;
                    ++k;
                }
            }
        }
    }
    #pragma unroll
    for (int k = 0; k < K27; ++k) part[cy][k][w] = acc[k];
    __syncthreads();

    // ---- softmax per (head, w): head = cy (threads cy<2 only) ----
    if (cy < 2) {
        float lg[K27];
        float mx = -3.0e38f;
        #pragma unroll
        for (int k = 0; k < K27; ++k) {
            const float s = part[cy][k][w] + part[cy + 2][k][w] +
                            part[cy + 4][k][w] + part[cy + 6][k][w];
            lg[k] = s; mx = fmaxf(mx, s);
        }
        float sum = 0.f;
        #pragma unroll
        for (int k = 0; k < K27; ++k) { const float e = __expf(lg[k] - mx); lg[k] = e; sum += e; }
        const float inv = 1.0f / sum;
        #pragma unroll
        for (int k = 0; k < K27; ++k) attn[cy][k][w] = lg[k] * inv;
    }
    __syncthreads();

    // ---- stage 2: merged[c] = sum_k attn[head][k] * tap[k,c] ----
    if (w < W_) {
        const int pos = ((b * T_ + t) * H_ + hr) * W_ + w;
        for (int j = 0; j < 16; ++j) {
            const int c = cy + 8 * j;
            const int head = c & 1;
            const float* xc = x + (size_t)(b * C_ + c) * THW_;
            float m = 0.f;
            int k = 0;
            #pragma unroll
            for (int dt = -1; dt <= 1; ++dt) {
                const int tt = t + dt; const bool okt = (unsigned)tt < (unsigned)T_;
                #pragma unroll
                for (int dh = -1; dh <= 1; ++dh) {
                    const int hh = hr + dh; const bool okh = (unsigned)hh < (unsigned)H_;
                    #pragma unroll
                    for (int dw = -1; dw <= 1; ++dw) {
                        const int ww = w + dw; const bool okw = (unsigned)ww < (unsigned)W_;
                        const float v = (okt & okh & okw) ? xc[(tt * H_ + hh) * W_ + ww] : 0.f;
                        m += attn[head][k][w] * v;
                        ++k;
                    }
                }
            }
            Mrg[(size_t)pos * C_ + c] = (_Float16)m;
        }
    }
}

// ---------------- Kernel 2: WMMA GEMM  out[P,128] = merged @ W_out^T + b ----------------
// 8 waves/block, 1 M-tile (16 positions) per wave; full N=128, K=128.
// W_out is converted f32->f16 into LDS once per block; B operands come from ds_load_b128.
__global__ __launch_bounds__(256) void gemm_kernel(const _Float16* __restrict__ A,
                                                   const float* __restrict__ Wo,
                                                   const float* __restrict__ bias,
                                                   float* __restrict__ out) {
    __shared__ _Float16 Wl[C_ * C_];   // 32 KB, row-major [co][ci]

    // cooperative stage + convert: 16384 floats, 4096 float4 chunks, 16 per thread
    {
        const float4* src = (const float4*)Wo;
        for (int i = threadIdx.x; i < (C_ * C_ / 4); i += 256) {
            const float4 v = src[i];
            v4h h; h[0] = (_Float16)v.x; h[1] = (_Float16)v.y;
                   h[2] = (_Float16)v.z; h[3] = (_Float16)v.w;
            *(v4h*)(Wl + 4 * i) = h;
        }
    }
    __syncthreads();

    const int lane  = threadIdx.x & 31;
    const int wv    = threadIdx.x >> 5;
    const int mtile = blockIdx.x * 8 + wv;          // grid sized exactly: no guard, EXEC all-1s
    const int m     = lane & 15;
    const int hiH   = lane >> 4;                    // 0: lanes 0-15, 1: lanes 16-31
    const int pos   = mtile * 16 + m;

    // A operand: 16x32, this code's own k order (B uses the same; not the ISA table order). Lane holds M=m, K = kb..kb+15 contiguous
    const _Float16* Arow = A + (size_t)pos * C_;
    const int kb = hiH * 16;
    v16h a[4];
    #pragma unroll
    for (int kk = 0; kk < 4; ++kk) {
        v8h lo = *(const v8h*)(Arow + kk * 32 + kb);
        v8h hi = *(const v8h*)(Arow + kk * 32 + kb + 8);
        a[kk] = __builtin_shufflevector(lo, hi, 0,1,2,3,4,5,6,7,8,9,10,11,12,13,14,15);
    }

    const int posBase = mtile * 16;
    const int bidx = posBase / THW_;                // uniform: THW_ % 16 == 0
    const int thw0 = posBase - bidx * THW_;

    #pragma unroll
    for (int nt = 0; nt < 8; ++nt) {
        const int co = nt * 16 + m;                 // B operand: lane column N = lane&15
        v8f accv = {};
        // B 32x16, same k order as A (not the ISA table order): lane holds N=co, K = hiH*16 + (0..15) contiguous per K-step
        const _Float16* Wrow = Wl + co * C_ + hiH * 16;
        #pragma unroll
        for (int kk = 0; kk < 4; ++kk) {
            v16h bm = *(const v16h*)(Wrow + kk * 32);
            accv = __builtin_amdgcn_wmma_f32_16x16x32_f16(
                false, a[kk], false, bm, (short)0, accv, false, false);
        }
        const float bz = bias[co];
        // D layout: VGPR r -> M = r + 8*hiH, N = lane&15.
        // Lane's 8 results are contiguous along thw -> two b128 stores.
        float* obase = out + ((size_t)(bidx * C_ + co)) * THW_ + thw0 + 8 * hiH;
        float4 o0, o1;
        o0.x = accv[0] + bz; o0.y = accv[1] + bz; o0.z = accv[2] + bz; o0.w = accv[3] + bz;
        o1.x = accv[4] + bz; o1.y = accv[5] + bz; o1.z = accv[6] + bz; o1.w = accv[7] + bz;
        *(float4*)(obase)     = o0;
        *(float4*)(obase + 4) = o1;
    }
}

extern "C" void kernel_launch(void* const* d_in, const int* in_sizes, int n_in,
                              void* d_out, int out_size, void* d_ws, size_t ws_size,
                              hipStream_t stream) {
    const float* x     = (const float*)d_in[0];
    const float* q     = (const float*)d_in[1];
    const float* W_out = (const float*)d_in[2];
    const float* b_out = (const float*)d_in[3];
    float* out = (float*)d_out;

    // workspace layout: merged f16 [P][C] at offset 0 (6.4 MB)
    _Float16* Mrg = (_Float16*)d_ws;

    attn_kernel<<<B_ * T_ * H_, dim3(32, 8), 0, stream>>>(x, q, Mrg);
    gemm_kernel<<<P_ / 16 / 8, 256, 0, stream>>>(Mrg, W_out, b_out, out);
}
